// Quantize_3753801417032
// MI455X (gfx1250) — matrix operand order repaired and checked by reading, not compiled, not run on hardware
//
#include <hip/hip_runtime.h>

// ---------------- types ----------------
typedef __attribute__((ext_vector_type(16))) __bf16          v16bf;
typedef __attribute__((ext_vector_type(8)))  float           v8f;
typedef __attribute__((ext_vector_type(4)))  unsigned int    v4u;
typedef __attribute__((ext_vector_type(2)))  float           v2f;
typedef __attribute__((ext_vector_type(4)))  float           v4f;
typedef __attribute__((ext_vector_type(4)))  unsigned short  v4us;

// ---------------- problem dims ----------------
#define BB_   32
#define C_    512
#define HW_   1024          // H*W
#define N_    32768         // B*H*W rows
#define K_    2048          // codebook entries
#define D_    512           // code dim
#define TROWS 128           // rows per block tile
#define TCOLS 128           // codes per code tile
#define DHALF 256           // staged D-chunk (two per code tile)
#define LDSP  520           // zt padded row stride (ushorts): 1040B, 16B-aligned, bank-spread
#define LDSW  264           // wt padded row stride (ushorts): 528B, 16B-aligned, bank-spread

// LDS layout (byte offsets inside dynamic shared alloc)
#define ZT_OFF   0
#define WN_OFF   133120                  // 128*520*2
#define WT_OFF   141312                  // + 2048*4
#define WT_BUFB  67584                   // 128*264*2 per buffer
#define RED_OFF  276480                  // WT_OFF + 2*WT_BUFB : cross-wave argmin scratch
#define LDS_TOT  278528                  // + 2*128*(4+4)

#define NSTAGE   (2 * (K_ / TCOLS))     // 32 stages of 128 codes x 256 dims

__device__ __forceinline__ unsigned short f2bf(float f) {
  unsigned u = __float_as_uint(f);
  u += 0x7FFFu + ((u >> 16) & 1u);          // round-to-nearest-even
  return (unsigned short)(u >> 16);
}

struct BfPair { v4u lo, hi; };
__device__ __forceinline__ v16bf mkfrag(const unsigned short* p0,
                                        const unsigned short* p1) {
  BfPair p;
  p.lo = *(const v4u*)p0;   // ds_load_b128
  p.hi = *(const v4u*)p1;   // ds_load_b128
  return __builtin_bit_cast(v16bf, p);
}

// Issue one 64KB weight stage as async LDS DMA: 16 x b128 per wave (x8 waves).
// GVS mode: mem = SADDR(64b) + VADDR(32b); VDST = LDS byte offset.
__device__ __forceinline__ void issue_stage(const unsigned short* __restrict__ wbf,
                                            int j, unsigned ldsbase, int tid) {
  const int ct = j >> 1, h = j & 1;
  const unsigned short* base = wbf + (size_t)ct * TCOLS * D_ + h * DHALF;
  const unsigned bufoff = ldsbase + WT_OFF + (unsigned)(j & 1) * WT_BUFB;
  const int lane = tid & 31, wave = tid >> 5;
  #pragma unroll
  for (int i = 0; i < 16; ++i) {
    const int f  = wave * 512 + i * 32 + lane;   // 4096 16B-chunks per stage
    const int kr = f >> 5;                       // code row 0..127
    const int dq = f & 31;                       // 16B chunk in 256-dim half
    const unsigned voff = (unsigned)(kr * 1024 + dq * 16);   // bytes from base
    const unsigned ldsa = bufoff + (unsigned)(kr * (LDSW * 2) + dq * 16);
    asm volatile("global_load_async_to_lds_b128 %0, %1, %2"
                 :: "v"(ldsa), "v"(voff), "s"(base) : "memory");
  }
}

// ---------------- kernel 1: weight -> bf16 + ||w||^2 ----------------
__global__ void __launch_bounds__(128)
vq_prep(const float* __restrict__ W, unsigned short* __restrict__ wbf,
        float* __restrict__ wnorm) {
  const int k = blockIdx.x, t = threadIdx.x;        // one block per code row
  v4f v = ((const v4f*)(W + (size_t)k * D_))[t];    // 128*4 = 512 elems
  v4us o = { f2bf(v.x), f2bf(v.y), f2bf(v.z), f2bf(v.w) };
  *(v4us*)(wbf + (size_t)k * D_ + t * 4) = o;
  float s = v.x*v.x + v.y*v.y + v.z*v.z + v.w*v.w;
  #pragma unroll
  for (int m = 16; m >= 1; m >>= 1) s += __shfl_xor(s, m, 32);
  __shared__ float red[4];
  if ((t & 31) == 0) red[t >> 5] = s;
  __syncthreads();
  if (t == 0) wnorm[k] = red[0] + red[1] + red[2] + red[3];
}

// ---------------- kernel 2: fused bf16-WMMA GEMM + row argmin ----------------
// block = 256 threads = 8 waves in a 4x2 (row x col) grid; tile = 128 rows x 2048 codes.
// Each wave: 32 rows x 64 cols -> 2 A-frags + 4 B-frags per K-step for 8 WMMAs
// (1.5 ds_load_b128 per WMMA). Weight tiles stream L2 -> LDS via async DMA,
// double buffered against WMMA.
// argmin_k(||x||^2 - 2 x.w_k + ||w_k||^2) == argmin_k(||w_k||^2 - 2 x.w_k).
__global__ void __launch_bounds__(256, 1)
vq_argmin_wmma(const float* __restrict__ z,
               const unsigned short* __restrict__ wbf,
               const float* __restrict__ wnorm,
               int* __restrict__ idx_out,
               float* __restrict__ idxf_out) {
  extern __shared__ char smem[];
  unsigned short* zt  = (unsigned short*)(smem + ZT_OFF);   // [128][520] bf16 z tile
  float*          wnL = (float*)(smem + WN_OFF);            // [2048] ||w||^2
  unsigned short* wtb = (unsigned short*)(smem + WT_OFF);   // 2 x [128][264] bf16
  float*          lv  = (float*)(smem + RED_OFF);           // [2][128] argmin val
  int*            li  = (int*)(smem + RED_OFF + 1024);      // [2][128] argmin idx
  const unsigned ldsbase = (unsigned)(uintptr_t)smem;       // LDS byte offset (flat low32)

  const int tid  = threadIdx.x;
  const int lane = tid & 31;
  const int wave = tid >> 5;
  const int mm   = lane & 15;
  const int hi   = lane >> 4;          // half-wave select (K-interleave per ISA layout)
  const int R0   = (wave >> 1) * 32;   // wave row group (4 groups of 32 rows)
  const int C0   = (wave & 1) * 64;    // wave col group (2 groups of 64 cols)
  const int blockR = blockIdx.x * TROWS;

  // --- stage z tile once: z is [B,C,H,W]; row n=(b,h,w), dim c at b*C*HW + c*HW + (n&1023)
  {
    const int bb  = blockR >> 10;      // 128 | 1024 so one batch per block
    const int rem = blockR & 1023;
    const float* zb = z + (size_t)bb * (C_ * HW_) + rem;
    const int rp = (tid & 63) * 2;     // row pair
    for (int c = tid >> 6; c < D_; c += 4) {
      v2f v = *(const v2f*)(zb + (size_t)c * HW_ + rp);   // coalesced along rows
      zt[(rp    ) * LDSP + c] = f2bf(v.x);
      zt[(rp + 1) * LDSP + c] = f2bf(v.y);
    }
    for (int i = tid; i < K_; i += 256) wnL[i] = wnorm[i];
  }

  float runv[2][8]; int runi[2][8];
  #pragma unroll
  for (int ri = 0; ri < 2; ++ri)
    #pragma unroll
    for (int r = 0; r < 8; ++r) { runv[ri][r] = 3.4e38f; runi[ri][r] = 0; }

  v8f acc[8];           // [ri*4 + ci]
  v8f zacc = {};

  issue_stage(wbf, 0, ldsbase, tid);   // prefetch first weight stage

  for (int j = 0; j < NSTAGE; ++j) {
    const int ct = j >> 1, h = j & 1;

    // stage j's DMA complete (per-wave, in-order), then make it visible block-wide.
    asm volatile("s_wait_asynccnt 0" ::: "memory");
    __syncthreads();

    // prefetch stage j+1 into the other buffer (its last reader finished at j-1,
    // which the barrier above ordered before us).
    if (j + 1 < NSTAGE) issue_stage(wbf, j + 1, ldsbase, tid);

    if (h == 0) {
      #pragma unroll
      for (int s = 0; s < 8; ++s) acc[s] = zacc;
    }

    const unsigned short* wbuf = wtb + (size_t)(j & 1) * (WT_BUFB / 2);
    const unsigned short* zr0 = zt + (R0 + mm) * LDSP + h * DHALF + hi * 8;
    const unsigned short* wr0 = wbuf + (C0 + mm) * LDSW + hi * 8;

    #pragma unroll
    for (int dsI = 0; dsI < 8; ++dsI) {
      const int d0 = dsI * 32;
      // A frags (16x32 bf16): lane row = mm; halves 0-7 = K[d0+8hi..+7], halves 8-15 = +16
      v16bf a[2];
      #pragma unroll
      for (int ri = 0; ri < 2; ++ri) {
        const unsigned short* p = zr0 + ri * (16 * LDSP) + d0;
        a[ri] = mkfrag(p, p + 16);
      }
      // B frags (32x16 bf16): lane col = mm; halves 0-7 = K[d0+8hi..+7], halves 8-15 = +16
      v16bf b[4];
      #pragma unroll
      for (int ci = 0; ci < 4; ++ci) {
        const unsigned short* q = wr0 + ci * (16 * LDSW) + d0;
        b[ci] = mkfrag(q, q + 16);
      }
      #pragma unroll
      for (int ri = 0; ri < 2; ++ri)
        #pragma unroll
        for (int ci = 0; ci < 4; ++ci)
          acc[ri * 4 + ci] = __builtin_amdgcn_wmma_f32_16x16x32_bf16(
              false, a[ri], false, b[ci], (short)0, acc[ri * 4 + ci], false, false);
    }

    if (h == 1) {
      // fold this code tile into per-lane running argmin (cols in increasing index)
      #pragma unroll
      for (int ci = 0; ci < 4; ++ci) {
        const int   coln = ct * TCOLS + C0 + ci * 16 + mm;
        const float wn   = wnL[coln];
        #pragma unroll
        for (int ri = 0; ri < 2; ++ri)
          #pragma unroll
          for (int r = 0; r < 8; ++r) {
            const float dist = fmaf(-2.0f, acc[ri * 4 + ci][r], wn);
            if (dist < runv[ri][r]) { runv[ri][r] = dist; runi[ri][r] = coln; }
          }
      }
    }
  }

  // --- butterfly min across the 16 lanes of each half, then cross-wave combine.
  // C/D layout: lanes 0-15 hold row r, lanes 16-31 row r+8 (per 16-row frag).
  #pragma unroll
  for (int ri = 0; ri < 2; ++ri)
    #pragma unroll
    for (int r = 0; r < 8; ++r) {
      float v = runv[ri][r]; int i = runi[ri][r];
      #pragma unroll
      for (int m = 1; m < 16; m <<= 1) {
        const float ov = __shfl_xor(v, m, 32);
        const int   oi = __shfl_xor(i, m, 32);
        if (ov < v || (ov == v && oi < i)) { v = ov; i = oi; }  // ties -> lowest index
      }
      if (mm == 0) {
        const int row = R0 + ri * 16 + hi * 8 + r;    // 0..127 within block tile
        lv[(wave & 1) * 128 + row] = v;
        li[(wave & 1) * 128 + row] = i;
      }
    }
  __syncthreads();
  if (tid < TROWS) {
    const float v0 = lv[tid], v1 = lv[128 + tid];
    const int   i0 = li[tid], i1 = li[128 + tid];
    const bool  t1 = (v1 < v0) || (v1 == v0 && i1 < i0);
    const int   i  = t1 ? i1 : i0;
    const int   gr = blockR + tid;
    idx_out[gr]  = i;
    idxf_out[gr] = (float)i;
  }
}

// ---------------- kernel 3: gather + straight-through writeback ----------------
__global__ void __launch_bounds__(256)
vq_gather(const float* __restrict__ z, const float* __restrict__ W,
          const int* __restrict__ idx, float* __restrict__ outq,
          float* __restrict__ outst) {
  const size_t e = (size_t)blockIdx.x * 256 + threadIdx.x;  // element of [B,C,H,W]
  const int pix = (int)(e & 1023);
  const int c   = (int)((e >> 10) & 511);
  const int b   = (int)(e >> 19);
  const int n   = (b << 10) | pix;
  const int k   = idx[n];                      // coalesced in n
  const float q  = W[(size_t)k * D_ + c];      // L2-resident gather
  const float zv = z[e];
  outq[e]  = q;
  outst[e] = (q - zv) + zv;                    // match reference fp arithmetic
}

// ---------------- launcher ----------------
extern "C" void kernel_launch(void* const* d_in, const int* in_sizes, int n_in,
                              void* d_out, int out_size, void* d_ws, size_t ws_size,
                              hipStream_t stream) {
  const float* z = (const float*)d_in[0];   // [32,512,32,32] f32
  const float* W = (const float*)d_in[1];   // [2048,512] f32

  unsigned short* wbf  = (unsigned short*)d_ws;                              // 2 MB
  float* wnorm = (float*)((char*)d_ws + (size_t)K_ * D_ * 2);                // 8 KB
  int*   idxb  = (int*)((char*)d_ws + (size_t)K_ * D_ * 2 + (size_t)K_ * 4); // 128 KB

  float* outq   = (float*)d_out;                         // quantized  [B,C,H,W]
  float* outst  = outq + (size_t)BB_ * C_ * HW_;         // straight_through
  float* outidx = outq + 2 * (size_t)BB_ * C_ * HW_;     // indices (as float)

  hipFuncSetAttribute((const void*)vq_argmin_wmma,
                      hipFuncAttributeMaxDynamicSharedMemorySize, LDS_TOT);

  vq_prep<<<K_, 128, 0, stream>>>(W, wbf, wnorm);
  vq_argmin_wmma<<<N_ / TROWS, 256, LDS_TOT, stream>>>(z, wbf, wnorm, idxb, outidx);
  vq_gather<<<(BB_ * C_ * HW_) / 256, 256, 0, stream>>>(z, W, idxb, outq, outst);
}
